// INT8SimulatedMatmul_10943576670803
// MI455X (gfx1250) — 8-bit matrix operand order repaired and checked by reading, compiled, not run on hardware
//
#include <hip/hip_runtime.h>
#include <cstdint>

typedef __attribute__((ext_vector_type(8))) int v8i;

__global__ __launch_bounds__(256)
void quantize_rows_kernel(const float* __restrict__ src,
                          int8_t* __restrict__ q,
                          float* __restrict__ scale,
                          int K)
{
    const int row = blockIdx.x;
    const float* r = src + (size_t)row * K;

    float amax = 0.0f;
    for (int i = threadIdx.x; i < K; i += blockDim.x)
        amax = fmaxf(amax, fabsf(r[i]));

    #pragma unroll
    for (int off = 16; off > 0; off >>= 1)
        amax = fmaxf(amax, __shfl_xor(amax, off, 32));

    __shared__ float red[8];
    const int wid = threadIdx.x >> 5;
    if ((threadIdx.x & 31) == 0) red[wid] = amax;
    __syncthreads();
    if (threadIdx.x < 32) {
        float v = (threadIdx.x < (blockDim.x >> 5)) ? red[threadIdx.x] : 0.0f;
        #pragma unroll
        for (int off = 4; off > 0; off >>= 1)
            v = fmaxf(v, __shfl_xor(v, off, 32));
        if (threadIdx.x == 0) red[0] = v;
    }
    __syncthreads();

    amax = fmaxf(red[0], 1e-8f);
    const float s   = amax * (1.0f / 127.0f);
    const float inv = 127.0f / amax;
    if (threadIdx.x == 0) scale[row] = s;

    uint32_t* qrow = (uint32_t*)(q + (size_t)row * K);
    for (int i = threadIdx.x * 4; i < K; i += blockDim.x * 4) {
        float4 v = *(const float4*)(r + i);
        int q0 = __float2int_rn(v.x * inv);
        int q1 = __float2int_rn(v.y * inv);
        int q2 = __float2int_rn(v.z * inv);
        int q3 = __float2int_rn(v.w * inv);
        q0 = max(-128, min(127, q0));
        q1 = max(-128, min(127, q1));
        q2 = max(-128, min(127, q2));
        q3 = max(-128, min(127, q3));
        uint32_t packed = (uint32_t)(q0 & 0xff)
                        | ((uint32_t)(q1 & 0xff) << 8)
                        | ((uint32_t)(q2 & 0xff) << 16)
                        | ((uint32_t)(q3 & 0xff) << 24);
        qrow[i >> 2] = packed;
    }
}

__global__ __launch_bounds__(128)
void int8_gemm_wmma_kernel(const int8_t* __restrict__ A8,
                           const int8_t* __restrict__ B8,
                           const float* __restrict__ aScale,
                           const float* __restrict__ bScale,
                           float* __restrict__ out,
                           int M, int N, int K)
{
    const int lane  = threadIdx.x & 31;
    const int wave  = threadIdx.x >> 5;
    const int half  = lane >> 4;
    const int l16   = lane & 15;

    const int waveM = (wave >> 1) * 64;
    const int waveN = (wave & 1)  * 64;
    const int blockM = blockIdx.y * 128;
    const int blockN = blockIdx.x * 128;

    v8i acc[4][4] = {};

    const int* arow[4];
    const int* brow[4];
    #pragma unroll
    for (int mi = 0; mi < 4; ++mi)
        arow[mi] = (const int*)(A8 + (size_t)(blockM + waveM + mi * 16 + l16) * K);
    #pragma unroll
    for (int ni = 0; ni < 4; ++ni)
        brow[ni] = (const int*)(B8 + (size_t)(blockN + waveN + ni * 16 + l16) * K);

    int aoff[8], boff[8];
    #pragma unroll
    for (int d = 0; d < 8; ++d) {
        aoff[d] = (((d & 3) * 4) +                         half * 16 + ((d >> 2) * 32)) >> 2;
        boff[d] = (((d & 3) * 4) +                         half * 16 + ((d >> 2) * 32)) >> 2;
    }

    for (int k0 = 0; k0 < K; k0 += 64) {
        const int kd = k0 >> 2;
        v8i a[4], b[4];
        #pragma unroll
        for (int mi = 0; mi < 4; ++mi) {
            #pragma unroll
            for (int d = 0; d < 8; ++d)
                a[mi][d] = arow[mi][kd + aoff[d]];
        }
        #pragma unroll
        for (int ni = 0; ni < 4; ++ni) {
            #pragma unroll
            for (int d = 0; d < 8; ++d)
                b[ni][d] = brow[ni][kd + boff[d]];
        }
        #pragma unroll
        for (int mi = 0; mi < 4; ++mi) {
            #pragma unroll
            for (int ni = 0; ni < 4; ++ni) {
                acc[mi][ni] = __builtin_amdgcn_wmma_i32_16x16x64_iu8(
                    true, a[mi], true, b[ni], acc[mi][ni], false, false);
            }
        }
    }

    #pragma unroll
    for (int mi = 0; mi < 4; ++mi) {
        #pragma unroll
        for (int r = 0; r < 8; ++r) {
            const int m = blockM + waveM + mi * 16 + r + half * 8;
            const float as = aScale[m];
            #pragma unroll
            for (int ni = 0; ni < 4; ++ni) {
                const int n = blockN + waveN + ni * 16 + l16;
                out[(size_t)m * N + n] = (float)acc[mi][ni][r] * as * bScale[n];
            }
        }
    }
}

extern "C" void kernel_launch(void* const* d_in, const int* in_sizes, int n_in,
                              void* d_out, int out_size, void* d_ws, size_t ws_size,
                              hipStream_t stream)
{
    const float* x = (const float*)d_in[0];
    const float* w = (const float*)d_in[1];

    const int K = 4096;
    const int M = in_sizes[0] / K;
    const int N = in_sizes[1] / K;

    int8_t* x8 = (int8_t*)d_ws;
    int8_t* w8 = x8 + (size_t)M * K;
    float*  xs = (float*)(w8 + (size_t)N * K);
    float*  wsc = xs + M;

    quantize_rows_kernel<<<M, 256, 0, stream>>>(x, x8, xs, K);
    quantize_rows_kernel<<<N, 256, 0, stream>>>(w, w8, wsc, K);

    dim3 grid(N / 128, M / 128);
    int8_gemm_wmma_kernel<<<grid, 128, 0, stream>>>(x8, w8, xs, wsc,
                                                    (float*)d_out, M, N, K);
}
